// LAVAMemory_26422638805504
// MI455X (gfx1250) — matrix operand order repaired and checked by reading, not compiled, not run on hardware
//
#include <hip/hip_runtime.h>
#include <math.h>

typedef unsigned short u16;
typedef unsigned int   u32;
typedef __attribute__((ext_vector_type(16))) __bf16 v16bf;
typedef __attribute__((ext_vector_type(8)))  float  v8f;

#define HIDDEN 2048
#define SLOTS  16384
#define NROWS  8192     // BATCH * SEQ
#define BM 64
#define BN 128
#define BK 32
#define LDA 48          // u16 stride for A tile in LDS (96B rows, 16B aligned slots)
#define LDB 48
#define SCHUNKS (SLOTS / BN)   // 128

// ---------- bf16 helpers ----------
__device__ __forceinline__ u16 f2bf(float f) {
  u32 u = __float_as_uint(f);
  u += 0x7FFFu + ((u >> 16) & 1u);      // round-to-nearest-even
  return (u16)(u >> 16);
}
__device__ __forceinline__ float bf2f(u16 h) {
  return __uint_as_float(((u32)h) << 16);
}

// ---------- CDNA5 async global->LDS copy (16B per lane, ASYNCcnt-tracked) ----------
// lds_off: wave-relative LDS byte address (low 32 bits of generic pointer).
__device__ __forceinline__ void async_ld16(u32 lds_off, const void* gaddr) {
  asm volatile("global_load_async_to_lds_b128 %0, %1, off"
               :: "v"(lds_off), "v"(gaddr) : "memory");
}
__device__ __forceinline__ void wait_async0() {
  asm volatile("s_wait_asynccnt 0x0" ::: "memory");
}

union FragU { v16bf v; uint4 q[2]; };

// A fragment (16x32 bf16, MxK). Lane: m=L&15, kbase=(L>>4)*8,
// VGPR0-3 = K[kbase..kbase+7], VGPR4-7 = K[kbase+16..kbase+23].
__device__ __forceinline__ v16bf frag_a_lds(const u16* As, int m, int kb) {
  FragU u;
  u.q[0] = *(const uint4*)(As + m * LDA + kb);
  u.q[1] = *(const uint4*)(As + m * LDA + kb + 16);
  return u.v;
}
// B fragment (32x16 bf16, KxN). Lane n=L&15; kbase=(L>>4)*8: VGPR0-3 = K[kbase..kbase+7], VGPR4-7 = K[kbase+16..kbase+23].
__device__ __forceinline__ v16bf frag_b_lds(const u16* Bs, int n, int kb) {
  FragU u;
  u.q[0] = *(const uint4*)(Bs + n * LDB + kb / 2);
  u.q[1] = *(const uint4*)(Bs + n * LDB + kb / 2 + 16);
  return u.v;
}

// ---------- cast f32 -> bf16 (4-wide) ----------
__global__ void cast_bf16_v4(const float* __restrict__ in, u16* __restrict__ out, size_t n4) {
  size_t i = (size_t)blockIdx.x * blockDim.x + threadIdx.x;
  size_t stride = (size_t)gridDim.x * blockDim.x;
  for (; i < n4; i += stride) {
    float4 v = ((const float4*)in)[i];
    uint2 o;
    o.x = ((u32)f2bf(v.y) << 16) | (u32)f2bf(v.x);
    o.y = ((u32)f2bf(v.w) << 16) | (u32)f2bf(v.z);
    ((uint2*)out)[i] = o;
  }
}

// ---------- row L2-normalize: f32 in -> bf16 out ----------
__global__ __launch_bounds__(256)
void rownorm_f32(const float* __restrict__ in, u16* __restrict__ out, float eps, int cols) {
  __shared__ float red[256];
  const int row = blockIdx.x, t = threadIdx.x;
  const float* r = in + (size_t)row * cols;
  float ss = 0.f;
  for (int c = t; c < cols; c += 256) { float v = r[c]; ss += v * v; }
  red[t] = ss; __syncthreads();
  for (int s = 128; s > 0; s >>= 1) { if (t < s) red[t] += red[t + s]; __syncthreads(); }
  float inv = 1.0f / fmaxf(sqrtf(red[0]), eps);
  for (int c = t; c < cols; c += 256) out[(size_t)row * cols + c] = f2bf(r[c] * inv);
}

// ---------- row L2-normalize: bf16 in -> bf16 out ----------
__global__ __launch_bounds__(256)
void rownorm_bf16(const u16* __restrict__ in, u16* __restrict__ out, float eps, int cols) {
  __shared__ float red[256];
  const int row = blockIdx.x, t = threadIdx.x;
  const u16* r = in + (size_t)row * cols;
  float ss = 0.f;
  for (int c = t; c < cols; c += 256) { float v = bf2f(r[c]); ss += v * v; }
  red[t] = ss; __syncthreads();
  for (int s = 128; s > 0; s >>= 1) { if (t < s) red[t] += red[t + s]; __syncthreads(); }
  float inv = 1.0f / fmaxf(sqrtf(red[0]), eps);
  for (int c = t; c < cols; c += 256) out[(size_t)row * cols + c] = f2bf(bf2f(r[c]) * inv);
}

// ---------- generic C = A(MxK) * B(NxK)^T, bf16 in, f32 or bf16 out ----------
// Double-buffered LDS tiles filled by async global->LDS; one barrier per K-step.
__global__ __launch_bounds__(256)
void gemm_bf16_tn(const u16* __restrict__ A, const u16* __restrict__ B,
                  float* __restrict__ Cf, u16* __restrict__ Cbf,
                  int M, int N, int K) {
  __shared__ __align__(16) u16 As[2][BM * LDA];
  __shared__ __align__(16) u16 Bs[2][BN * LDB];
  const int t = threadIdx.x;
  const int lane = t & 31, w = t >> 5;
  const int wr = w >> 2, wc = w & 3;           // wave grid 2x4 -> 32x32 per wave
  const int lm = lane & 15, lh = lane >> 4;
  const int mbase = blockIdx.x * BM;
  const int nbase = blockIdx.y * BN;

  const int stA = (t >> 2) * LDA + (t & 3) * 8;     // this thread's A slot
  const int stB = (t >> 1) * LDB + (t & 1) * 16;    // this thread's B slot
  const u32 ldsA[2] = { (u32)(size_t)&As[0][stA], (u32)(size_t)&As[1][stA] };
  const u32 ldsB[2] = { (u32)(size_t)&Bs[0][stB], (u32)(size_t)&Bs[1][stB] };
  const u16* gA = A + (size_t)(mbase + (t >> 2)) * K + (t & 3) * 8;
  const u16* gB = B + (size_t)(nbase + (t >> 1)) * K + (t & 1) * 16;

  v8f c[2][2] = {};
  // prologue: stage tile 0
  async_ld16(ldsA[0], gA);
  async_ld16(ldsB[0], gB);
  async_ld16(ldsB[0] + 16, gB + 8);

  int buf = 0;
  for (int k0 = 0; k0 < K; k0 += BK, buf ^= 1) {
    wait_async0();           // my async writes for tile(buf) have landed
    __syncthreads();         // everyone's have; previous compute fully done
    if (k0 + BK < K) {       // stage tile k0+BK into the other buffer
      async_ld16(ldsA[buf ^ 1], gA + k0 + BK);
      async_ld16(ldsB[buf ^ 1], gB + k0 + BK);
      async_ld16(ldsB[buf ^ 1] + 16, gB + k0 + BK + 8);
    }
    const u16* Ab = As[buf]; const u16* Bb = Bs[buf];
    v16bf a0 = frag_a_lds(Ab, wr * 32 + lm,      lh * 8);
    v16bf a1 = frag_a_lds(Ab, wr * 32 + 16 + lm, lh * 8);
    v16bf b0 = frag_b_lds(Bb, wc * 32 + lm,      lh * 16);
    v16bf b1 = frag_b_lds(Bb, wc * 32 + 16 + lm, lh * 16);
    c[0][0] = __builtin_amdgcn_wmma_f32_16x16x32_bf16(false, a0, false, b0, (short)0, c[0][0], false, false);
    c[0][1] = __builtin_amdgcn_wmma_f32_16x16x32_bf16(false, a0, false, b1, (short)0, c[0][1], false, false);
    c[1][0] = __builtin_amdgcn_wmma_f32_16x16x32_bf16(false, a1, false, b0, (short)0, c[1][0], false, false);
    c[1][1] = __builtin_amdgcn_wmma_f32_16x16x32_bf16(false, a1, false, b1, (short)0, c[1][1], false, false);
  }
#pragma unroll
  for (int mi = 0; mi < 2; ++mi)
#pragma unroll
    for (int ni = 0; ni < 2; ++ni)
#pragma unroll
      for (int r = 0; r < 8; ++r) {
        int row = mbase + wr * 32 + mi * 16 + lh * 8 + r;   // M = r + 8*(L>>4)
        int col = nbase + wc * 32 + ni * 16 + lm;           // N = L&15
        float v = c[mi][ni][r];
        if (Cbf) Cbf[(size_t)row * N + col] = f2bf(v);
        else     Cf [(size_t)row * N + col] = v;
      }
}

// ---------- fused scores GEMM + running top-4 + softmax ----------
// LDS pool: double-buffered staging (36KB) aliased with the 32KB score tile
// (phases separated by barriers; asyncs fully drained at K-loop exit).
__global__ __launch_bounds__(256)
void scores_topk(const u16* __restrict__ Q,    // [NROWS][HIDDEN] bf16, normalized
                 const u16* __restrict__ AD,   // [SLOTS][HIDDEN] bf16, normalized
                 float* __restrict__ attn,     // [NROWS][4] softmaxed weights
                 int* __restrict__ outidx) {   // [NROWS][4]
  __shared__ __align__(16) char pool[45056];
  u16* As0 = (u16*)pool;                  //  6KB
  u16* As1 = (u16*)(pool + 6144);         //  6KB
  u16* Bs0 = (u16*)(pool + 12288);        // 12KB
  u16* Bs1 = (u16*)(pool + 24576);        // 12KB
  float* Sc = (float*)pool;               // 32KB (aliases staging)
  float* mv = (float*)(pool + 36864);     //  4KB
  int*  mxi = (int*)(pool + 40960);       //  4KB

  const int t = threadIdx.x;
  const int lane = t & 31, w = t >> 5;
  const int wr = w >> 2, wc = w & 3;
  const int lm = lane & 15, lh = lane >> 4;
  const int mbase = blockIdx.x * BM;

  const int stA = (t >> 2) * LDA + (t & 3) * 8;
  const int stB = (t >> 1) * LDB + (t & 1) * 16;
  const u32 ldsA[2] = { (u32)(size_t)(As0 + stA), (u32)(size_t)(As1 + stA) };
  const u32 ldsB[2] = { (u32)(size_t)(Bs0 + stB), (u32)(size_t)(Bs1 + stB) };
  const u16* Abase[2] = { As0, As1 };
  const u16* Bbase[2] = { Bs0, Bs1 };
  const u16* gA = Q + (size_t)(mbase + (t >> 2)) * HIDDEN + (t & 3) * 8;

  const int srow  = t >> 2;          // 0..63 : row this thread scans
  const int scol0 = (t & 3) * 32;    // 32-col strip per thread

  float v0 = -1e30f, v1 = -1e30f, v2 = -1e30f, v3 = -1e30f;
  int   i0 = 0, i1 = 0, i2 = 0, i3 = 0;

  for (int ch = 0; ch < SCHUNKS; ++ch) {
    const int nbase = ch * BN;
    const u16* gB = AD + (size_t)(nbase + (t >> 1)) * HIDDEN + (t & 1) * 16;
    v8f c[2][2] = {};
    // prologue for this chunk (previous chunk's scan barrier already passed)
    async_ld16(ldsA[0], gA);
    async_ld16(ldsB[0], gB);
    async_ld16(ldsB[0] + 16, gB + 8);
    int buf = 0;
    for (int k0 = 0; k0 < HIDDEN; k0 += BK, buf ^= 1) {
      wait_async0();
      __syncthreads();
      if (k0 + BK < HIDDEN) {
        async_ld16(ldsA[buf ^ 1], gA + k0 + BK);
        async_ld16(ldsB[buf ^ 1], gB + k0 + BK);
        async_ld16(ldsB[buf ^ 1] + 16, gB + k0 + BK + 8);
      }
      const u16* Ab = Abase[buf]; const u16* Bb = Bbase[buf];
      v16bf a0 = frag_a_lds(Ab, wr * 32 + lm,      lh * 8);
      v16bf a1 = frag_a_lds(Ab, wr * 32 + 16 + lm, lh * 8);
      v16bf b0 = frag_b_lds(Bb, wc * 32 + lm,      lh * 16);
      v16bf b1 = frag_b_lds(Bb, wc * 32 + 16 + lm, lh * 16);
      c[0][0] = __builtin_amdgcn_wmma_f32_16x16x32_bf16(false, a0, false, b0, (short)0, c[0][0], false, false);
      c[0][1] = __builtin_amdgcn_wmma_f32_16x16x32_bf16(false, a0, false, b1, (short)0, c[0][1], false, false);
      c[1][0] = __builtin_amdgcn_wmma_f32_16x16x32_bf16(false, a1, false, b0, (short)0, c[1][0], false, false);
      c[1][1] = __builtin_amdgcn_wmma_f32_16x16x32_bf16(false, a1, false, b1, (short)0, c[1][1], false, false);
    }
    __syncthreads();   // all waves done reading staging before Sc dump (alias)
    // dump 64x128 fp32 score tile to LDS
#pragma unroll
    for (int mi = 0; mi < 2; ++mi)
#pragma unroll
      for (int ni = 0; ni < 2; ++ni)
#pragma unroll
        for (int r = 0; r < 8; ++r) {
          int row = wr * 32 + mi * 16 + lh * 8 + r;
          int col = wc * 32 + ni * 16 + lm;
          Sc[row * BN + col] = c[mi][ni][r];
        }
    __syncthreads();
    // running top-4 insertion (sorted descending)
#pragma unroll 4
    for (int j = 0; j < 32; ++j) {
      int col = scol0 + j;
      float s = Sc[srow * BN + col];
      int gi = nbase + col;
      if (s > v3) {
        if (s > v0)      { v3=v2;i3=i2; v2=v1;i2=i1; v1=v0;i1=i0; v0=s;i0=gi; }
        else if (s > v1) { v3=v2;i3=i2; v2=v1;i2=i1; v1=s;i1=gi; }
        else if (s > v2) { v3=v2;i3=i2; v2=s;i2=gi; }
        else             { v3=s;i3=gi; }
      }
    }
    __syncthreads();   // scan complete before next chunk's prologue overwrites Sc
  }

  // merge the 4 per-row partial top-4 lists
  int mb = srow * 16 + (t & 3) * 4;
  mv[mb + 0] = v0; mxi[mb + 0] = i0;
  mv[mb + 1] = v1; mxi[mb + 1] = i1;
  mv[mb + 2] = v2; mxi[mb + 2] = i2;
  mv[mb + 3] = v3; mxi[mb + 3] = i3;
  __syncthreads();
  if (t < BM) {
    float cv[16]; int ci[16];
#pragma unroll
    for (int j = 0; j < 16; ++j) { cv[j] = mv[t * 16 + j]; ci[j] = mxi[t * 16 + j]; }
    float bv[4]; int bi[4];
#pragma unroll
    for (int k = 0; k < 4; ++k) {
      int best = 0;
      for (int j = 1; j < 16; ++j) if (cv[j] > cv[best]) best = j;
      bv[k] = cv[best]; bi[k] = ci[best]; cv[best] = -1e38f;
    }
    float m0 = bv[0];
    float e0 = __expf(bv[0] - m0), e1 = __expf(bv[1] - m0);
    float e2 = __expf(bv[2] - m0), e3 = __expf(bv[3] - m0);
    float inv = 1.0f / (e0 + e1 + e2 + e3);
    int row = mbase + t;
    attn[row * 4 + 0] = e0 * inv; outidx[row * 4 + 0] = bi[0];
    attn[row * 4 + 1] = e1 * inv; outidx[row * 4 + 1] = bi[1];
    attn[row * 4 + 2] = e2 * inv; outidx[row * 4 + 2] = bi[2];
    attn[row * 4 + 3] = e3 * inv; outidx[row * 4 + 3] = bi[3];
  }
}

// ---------- mem[row] = sum_k attn[k] * contents[idx[k]]  ->  bf16 ----------
__global__ __launch_bounds__(256)
void gather_mem(const float* __restrict__ contents, const float* __restrict__ attn,
                const int* __restrict__ idx, u16* __restrict__ mem) {
  const int row = blockIdx.x, t = threadIdx.x;
  float a0 = attn[row * 4 + 0], a1 = attn[row * 4 + 1];
  float a2 = attn[row * 4 + 2], a3 = attn[row * 4 + 3];
  const float4* c0 = (const float4*)(contents + (size_t)idx[row * 4 + 0] * HIDDEN);
  const float4* c1 = (const float4*)(contents + (size_t)idx[row * 4 + 1] * HIDDEN);
  const float4* c2 = (const float4*)(contents + (size_t)idx[row * 4 + 2] * HIDDEN);
  const float4* c3 = (const float4*)(contents + (size_t)idx[row * 4 + 3] * HIDDEN);
  uint2* o = (uint2*)(mem + (size_t)row * HIDDEN);
#pragma unroll
  for (int i = 0; i < HIDDEN / 4 / 256; ++i) {   // 2 iterations
    int p = t + i * 256;
    float4 x0 = c0[p], x1 = c1[p], x2 = c2[p], x3 = c3[p];
    float sx = a0 * x0.x + a1 * x1.x + a2 * x2.x + a3 * x3.x;
    float sy = a0 * x0.y + a1 * x1.y + a2 * x2.y + a3 * x3.y;
    float sz = a0 * x0.z + a1 * x1.z + a2 * x2.z + a3 * x3.z;
    float sw = a0 * x0.w + a1 * x1.w + a2 * x2.w + a3 * x3.w;
    uint2 pk;
    pk.x = ((u32)f2bf(sy) << 16) | (u32)f2bf(sx);
    pk.y = ((u32)f2bf(sw) << 16) | (u32)f2bf(sz);
    o[p] = pk;
  }
}

extern "C" void kernel_launch(void* const* d_in, const int* in_sizes, int n_in,
                              void* d_out, int out_size, void* d_ws, size_t ws_size,
                              hipStream_t stream) {
  const float* x     = (const float*)d_in[0];
  const float* Wa    = (const float*)d_in[1];
  const float* Wr    = (const float*)d_in[2];
  const float* addrs = (const float*)d_in[3];
  const float* cont  = (const float*)d_in[4];
  float* out = (float*)d_out;

  char* ws = (char*)d_ws;
  size_t o = 0;
  u16* x_bf  = (u16*)(ws + o); o += (size_t)NROWS  * HIDDEN * 2;  // 32MB (reused as qn)
  u16* Wa_bf = (u16*)(ws + o); o += (size_t)HIDDEN * HIDDEN * 2;  // 8MB
  u16* Wr_bf = (u16*)(ws + o); o += (size_t)HIDDEN * HIDDEN * 2;  // 8MB
  u16* an_bf = (u16*)(ws + o); o += (size_t)SLOTS  * HIDDEN * 2;  // 64MB
  u16* q_bf  = (u16*)(ws + o); o += (size_t)NROWS  * HIDDEN * 2;  // 32MB (reused as mem)
  float* attn = (float*)(ws + o); o += (size_t)NROWS * 4 * sizeof(float);
  int*   topi = (int*)(ws + o);   o += (size_t)NROWS * 4 * sizeof(int);
  u16* qn_bf  = x_bf;   // x dead after q GEMM
  u16* mem_bf = q_bf;   // q dead after qnorm

  // 1) casts + address normalization
  cast_bf16_v4<<<2048, 256, 0, stream>>>(x,  x_bf,  (size_t)NROWS * HIDDEN / 4);
  cast_bf16_v4<<<512,  256, 0, stream>>>(Wa, Wa_bf, (size_t)HIDDEN * HIDDEN / 4);
  cast_bf16_v4<<<512,  256, 0, stream>>>(Wr, Wr_bf, (size_t)HIDDEN * HIDDEN / 4);
  rownorm_f32<<<SLOTS, 256, 0, stream>>>(addrs, an_bf, 1e-8f, HIDDEN);

  // 2) q = x @ Wa^T  (bf16 out)
  gemm_bf16_tn<<<dim3(NROWS / BM, HIDDEN / BN), 256, 0, stream>>>(
      x_bf, Wa_bf, nullptr, q_bf, NROWS, HIDDEN, HIDDEN);

  // 3) q-normalize -> qn
  rownorm_bf16<<<NROWS, 256, 0, stream>>>(q_bf, qn_bf, 1e-6f, HIDDEN);

  // 4) fused cosine-score GEMM + top-4 + softmax
  scores_topk<<<NROWS / BM, 256, 0, stream>>>(qn_bf, an_bf, attn, topi);

  // 5) weighted gather of contents -> mem (bf16)
  gather_mem<<<NROWS, 256, 0, stream>>>(cont, attn, topi, mem_bf);

  // 6) out = mem @ Wr^T  (fp32)
  gemm_bf16_tn<<<dim3(NROWS / BM, HIDDEN / BN), 256, 0, stream>>>(
      mem_bf, Wr_bf, out, nullptr, NROWS, HIDDEN, HIDDEN);
}
